// MaxKSAGEConv_45938970198373
// MI455X (gfx1250) — compile-verified, matrix operand order checked by reading, not run on hardware
//
#include <hip/hip_runtime.h>
#include <hip/hip_bf16.h>

// ---------------------------------------------------------------------------
// MaxK-SAGEConv for MI455X (gfx1250, wave32, WMMA bf16 path)
//   out = feat @ W_self^T  +  CSR_spmm( topk32(feat @ W_neigh^T + b) )
// N=50000, IN=OUT=256, E=800000, K=32
// ---------------------------------------------------------------------------

typedef __bf16 bf16_t;
typedef __attribute__((ext_vector_type(16))) __bf16 v16bf;
typedef __attribute__((ext_vector_type(8)))  __bf16 v8bf;
typedef __attribute__((ext_vector_type(8)))  float  v8f;

#define DIM     256      // IN == OUT
#define TOPK    32
#define LDSPAD  264      // 256 + 8 pad: breaks 512B bank-conflict stride

static __device__ __forceinline__ bf16_t f2bf(float f) { return (bf16_t)f; }

// ---------------------------------------------------------------------------
// Kernel 1: pack W (256x256 f32, row-major [out][in]) into bf16 WMMA B
// fragments. B[k][n] = W[n][k]. Fragment layout per CDNA5 ISA 7.12.2
// (16-bit 32x16 B): lane l holds column n = l&15; element i of the v16bf
// holds K = kt*32 + (i<8 ? i : i+8) + ((l>=16) ? 8 : 0).
// Packed index: (((kt*16 + ct)*32 + lane)*16 + i)  -> contiguous 32B/lane.
// ---------------------------------------------------------------------------
__global__ __launch_bounds__(256) void pack_w_kernel(
    const float* __restrict__ Ws, const float* __restrict__ Wn,
    bf16_t* __restrict__ pWs, bf16_t* __restrict__ pWn)
{
    int t = blockIdx.x * 256 + threadIdx.x;     // [0, 2*65536)
    int w    = t >> 16;
    int rem  = t & 65535;
    int i    = rem & 15;
    int lane = (rem >> 4) & 31;
    int ctkt = rem >> 9;
    int ct   = ctkt & 15;
    int kt   = ctkt >> 4;
    int col  = ct * 16 + (lane & 15);
    int K    = kt * 32 + (i < 8 ? i : i + 8) + ((lane >> 4) & 1) * 8;
    const float* W = w ? Wn : Ws;
    bf16_t*      P = w ? pWn : pWs;
    P[rem] = f2bf(W[col * DIM + K]);
}

// ---------------------------------------------------------------------------
// Kernel 2: fused dual-GEMM + bias + top-32 compaction.
// Block = 256 threads (8 waves) handles a 16-row slab.
//   Phase A: stage feat tile (16x256) f32->bf16 into LDS.
//   Phase B: wave w owns column tiles ct0=2w, ct1=2w+1; 8 K-steps of
//            v_wmma_f32_16x16x32_bf16 for both W_self and W_neigh.
//   Phase C: h_self -> d_out; feat_neigh+bias -> LDS (f32).
//   Phase D: each wave argmax-selects top-32 of 2 rows, emits compact
//            (vals, cols) to workspace.
// ---------------------------------------------------------------------------
__global__ __launch_bounds__(256) void gemm_maxk_kernel(
    const float*  __restrict__ feat,
    const bf16_t* __restrict__ pWs,
    const bf16_t* __restrict__ pWn,
    const float*  __restrict__ bias,
    float*        __restrict__ out,     // h_self
    float*        __restrict__ valsC,   // [N][32]
    int*          __restrict__ colsC)   // [N][32]
{
    __shared__ bf16_t sA[16 * LDSPAD];  // feat tile, bf16
    __shared__ float  sF[16 * LDSPAD];  // feat_neigh tile, f32

    const int t    = threadIdx.x;
    const int lane = t & 31;
    const int wave = t >> 5;
    const int row0 = blockIdx.x * 16;

    // ---- Phase A: cooperative load + convert -------------------------------
    {
        int r = t >> 4, seg = t & 15;               // 16 floats per thread
        const float4* src = (const float4*)(feat + (row0 + r) * DIM + seg * 16);
        bf16_t* dst = &sA[r * LDSPAD + seg * 16];
        #pragma unroll
        for (int q = 0; q < 4; ++q) {
            float4 f = src[q];
            dst[q * 4 + 0] = f2bf(f.x);
            dst[q * 4 + 1] = f2bf(f.y);
            dst[q * 4 + 2] = f2bf(f.z);
            dst[q * 4 + 3] = f2bf(f.w);
        }
    }
    __syncthreads();

    // ---- Phase B: WMMA main loop ------------------------------------------
    const int lanehi = lane >> 4;          // 0 or 1
    const int kofs   = lanehi * 8;         // A-frag K sub-offset per ISA layout
    const int ct0    = wave * 2;
    const int ct1    = ct0 + 1;
    const v16bf* Bs  = (const v16bf*)pWs;
    const v16bf* Bn  = (const v16bf*)pWn;

    v8f as0 = {}, as1 = {}, an0 = {}, an1 = {};

    for (int kt = 0; kt < 8; ++kt) {
        // A fragment: row = lane&15; elems 0..7 at K=kt*32+kofs, 8..15 at +16
        const v8bf* pa = (const v8bf*)&sA[(lane & 15) * LDSPAD + kt * 32 + kofs];
        v8bf alo = pa[0];
        v8bf ahi = pa[2];                  // +16 bf16 elements
        v16bf a = __builtin_shufflevector(alo, ahi,
                     0,1,2,3,4,5,6,7,8,9,10,11,12,13,14,15);

        v16bf bs0 = Bs[(kt * 16 + ct0) * 32 + lane];
        v16bf bs1 = Bs[(kt * 16 + ct1) * 32 + lane];
        v16bf bn0 = Bn[(kt * 16 + ct0) * 32 + lane];
        v16bf bn1 = Bn[(kt * 16 + ct1) * 32 + lane];

        as0 = __builtin_amdgcn_wmma_f32_16x16x32_bf16(false, a, false, bs0, (short)0, as0, false, false);
        as1 = __builtin_amdgcn_wmma_f32_16x16x32_bf16(false, a, false, bs1, (short)0, as1, false, false);
        an0 = __builtin_amdgcn_wmma_f32_16x16x32_bf16(false, a, false, bn0, (short)0, an0, false, false);
        an1 = __builtin_amdgcn_wmma_f32_16x16x32_bf16(false, a, false, bn1, (short)0, an1, false, false);
    }

    // ---- Phase C: epilogue. C/D layout: lane l -> col = l&15,
    // VGPR v -> row = v + 8*(l>=16).
    {
        int col = ct0 * 16 + (lane & 15);
        float b = bias[col];
        float* o = out + (row0 + lanehi * 8) * DIM + col;
        float* f = &sF[(lanehi * 8) * LDSPAD + col];
        #pragma unroll
        for (int v = 0; v < 8; ++v) { o[v * DIM] = as0[v]; f[v * LDSPAD] = an0[v] + b; }
    }
    {
        int col = ct1 * 16 + (lane & 15);
        float b = bias[col];
        float* o = out + (row0 + lanehi * 8) * DIM + col;
        float* f = &sF[(lanehi * 8) * LDSPAD + col];
        #pragma unroll
        for (int v = 0; v < 8; ++v) { o[v * DIM] = as1[v]; f[v * LDSPAD] = an1[v] + b; }
    }
    __syncthreads();

    // ---- Phase D: top-32 per row (wave-wide iterative argmax) -------------
    #pragma unroll
    for (int rr = 0; rr < 2; ++rr) {
        int r = wave * 2 + rr;
        float x[8];
        #pragma unroll
        for (int j = 0; j < 8; ++j) x[j] = sF[r * LDSPAD + lane + 32 * j];

        float myv = 0.0f; int myi = 0;
        for (int round = 0; round < TOPK; ++round) {
            float bv = x[0]; int bi = lane;
            #pragma unroll
            for (int j = 1; j < 8; ++j)
                if (x[j] > bv) { bv = x[j]; bi = lane + 32 * j; }
            // butterfly reduction: all lanes end with the global (max, argmin-idx)
            #pragma unroll
            for (int off = 16; off > 0; off >>= 1) {
                float ov = __shfl_xor(bv, off, 32);
                int   oi = __shfl_xor(bi, off, 32);
                if (ov > bv || (ov == bv && oi < bi)) { bv = ov; bi = oi; }
            }
            if (lane == round) { myv = bv; myi = bi; }
            if (lane == (bi & 31)) {        // owner retires the winner
                int slot = bi >> 5;
                #pragma unroll
                for (int j = 0; j < 8; ++j)
                    if (slot == j) x[j] = -__builtin_inff();
            }
        }
        int rg = row0 + r;
        valsC[rg * TOPK + lane] = myv;
        colsC[rg * TOPK + lane] = myi;
    }
}

// ---------------------------------------------------------------------------
// Kernel 3: CSR SpMM over the compact top-32 representation + final add.
// One wave per destination node; 256-float LDS accumulator per wave;
// ds_add_f32 atomics resolve in-wave column collisions.
// ---------------------------------------------------------------------------
__global__ __launch_bounds__(256) void spmm_kernel(
    const int*   __restrict__ indices,
    const int*   __restrict__ indptr,
    const float* __restrict__ valsC,
    const int*   __restrict__ colsC,
    float*       __restrict__ out,
    int n)
{
    __shared__ float acc[8 * DIM];
    const int lane = threadIdx.x & 31;
    const int wave = threadIdx.x >> 5;
    const int node = blockIdx.x * 8 + wave;
    if (node >= n) return;

    float* a = &acc[wave * DIM];
    #pragma unroll
    for (int j = 0; j < 8; ++j) a[lane + 32 * j] = 0.0f;

    const int s = indptr[node];
    const int e = indptr[node + 1];
    for (int ed = s; ed < e; ++ed) {
        int nb   = indices[ed];
        float v  = valsC[nb * TOPK + lane];
        int   c  = colsC[nb * TOPK + lane];
        atomicAdd(&a[c], v);                    // ds_add_f32 (no-return)
    }

    #pragma unroll
    for (int j = 0; j < 8; ++j) {
        int c = lane + 32 * j;
        out[node * DIM + c] += a[c];            // out already holds h_self
    }
}

// ---------------------------------------------------------------------------
extern "C" void kernel_launch(void* const* d_in, const int* in_sizes, int n_in,
                              void* d_out, int out_size, void* d_ws, size_t ws_size,
                              hipStream_t stream)
{
    const float* feat    = (const float*)d_in[0];
    const float* W_self  = (const float*)d_in[1];
    const float* W_neigh = (const float*)d_in[2];
    const float* b_neigh = (const float*)d_in[3];
    const int*   indices = (const int*)d_in[4];
    const int*   indptr  = (const int*)d_in[5];
    float*       out     = (float*)d_out;

    const int N = in_sizes[0] / DIM;            // 50000 (multiple of 16)

    // Workspace layout (all rewritten every launch):
    //   [0, 128K)        packed W_self  bf16 fragments
    //   [128K, 256K)     packed W_neigh bf16 fragments
    //   [256K, +N*32*4)  compact top-k values
    //   [..., +N*32*4)   compact top-k column indices
    char* ws = (char*)d_ws;
    bf16_t* pWs   = (bf16_t*)(ws);
    bf16_t* pWn   = (bf16_t*)(ws + 131072);
    float*  valsC = (float*)(ws + 262144);
    int*    colsC = (int*)  (ws + 262144 + (size_t)N * TOPK * sizeof(float));

    // 1) pre-swizzle weights into WMMA B-fragment order (bf16)
    pack_w_kernel<<<512, 256, 0, stream>>>(W_self, W_neigh, pWs, pWn);

    // 2) fused dual-GEMM + bias + top-32 compaction (16 rows / block)
    gemm_maxk_kernel<<<N / 16, 256, 0, stream>>>(feat, pWs, pWn, b_neigh,
                                                 out, valsC, colsC);

    // 3) sparse aggregation + final add (8 nodes / block, 1 wave / node)
    spmm_kernel<<<(N + 7) / 8, 256, 0, stream>>>(indices, indptr,
                                                 valsC, colsC, out, N);
}
